// FP4Linear_42975442764444
// MI455X (gfx1250) — matrix operand order repaired and checked by reading, not compiled, not run on hardware
//
#include <hip/hip_runtime.h>

// ---- CDNA5 (gfx1250) wave32 WMMA types ----
typedef __attribute__((ext_vector_type(16))) __bf16 v16bf;
typedef __attribute__((ext_vector_type(8)))  __bf16 v8bf;
typedef __attribute__((ext_vector_type(8)))  float  v8f;
typedef __attribute__((ext_vector_type(4)))  int    v4i;

// GEMM: out[M,N] = x[M,K] * (scale * Wq[N,K])^T + bias
//   M = B*S = 8192, N = D_OUT = 4096, K = D_IN = 4096
#define BM 128         // block rows
#define BN 256         // block cols (8 waves as 2x4 grid of 64x64 tiles)
#define BK 64          // one LDS stage = two WMMA K-chunks of 32
#define LDA 72         // padded LDS row (bf16 elems): 144B, 16B-aligned chunks
#define LDB 72

// ---------- CDNA5 async-copy-to-LDS path (ASYNCcnt DMA) ----------
#if __has_builtin(__builtin_amdgcn_global_load_async_to_lds_b128) && \
    __has_builtin(__builtin_amdgcn_s_wait_asynccnt)
#define HAVE_ASYNC_LDS 1
#else
#define HAVE_ASYNC_LDS 0
#endif

#if HAVE_ASYNC_LDS
__device__ __forceinline__ void async_cp_b128(const __bf16* g, __bf16* l) {
    __builtin_amdgcn_global_load_async_to_lds_b128(
        (__attribute__((address_space(1))) v4i*)(v4i*)(void*)const_cast<__bf16*>(g),
        (__attribute__((address_space(3))) v4i*)(v4i*)(void*)l,
        /*imm offset*/0, /*cpol*/0);
}
#endif

// =======================================================================
// Pre-pass 1: dequantize int4-code weights -> bf16 * scale  (once per call)
// =======================================================================
__launch_bounds__(256)
__global__ void cvt_w_bf16(const int* __restrict__ wq,
                           const float* __restrict__ scale_p,
                           __bf16* __restrict__ wb) {
    const float s = scale_p[0];
    const size_t i = ((size_t)blockIdx.x * 256 + threadIdx.x) * 8;
    int4 a = *(const int4*)(wq + i);
    int4 b = *(const int4*)(wq + i + 4);
    v8bf o;
    o[0] = (__bf16)((float)a.x * s); o[1] = (__bf16)((float)a.y * s);
    o[2] = (__bf16)((float)a.z * s); o[3] = (__bf16)((float)a.w * s);
    o[4] = (__bf16)((float)b.x * s); o[5] = (__bf16)((float)b.y * s);
    o[6] = (__bf16)((float)b.z * s); o[7] = (__bf16)((float)b.w * s);
    *(v8bf*)(wb + i) = o;
}

// =======================================================================
// Pre-pass 2: x f32 -> bf16  (once per call)
// =======================================================================
__launch_bounds__(256)
__global__ void cvt_x_bf16(const float* __restrict__ x,
                           __bf16* __restrict__ xb) {
    const size_t i = ((size_t)blockIdx.x * 256 + threadIdx.x) * 8;
    float4 a = *(const float4*)(x + i);
    float4 b = *(const float4*)(x + i + 4);
    v8bf o;
    o[0] = (__bf16)a.x; o[1] = (__bf16)a.y; o[2] = (__bf16)a.z; o[3] = (__bf16)a.w;
    o[4] = (__bf16)b.x; o[5] = (__bf16)b.y; o[6] = (__bf16)b.z; o[7] = (__bf16)b.w;
    *(v8bf*)(xb + i) = o;
}

// =======================================================================
// Main GEMM: bf16 x bf16 -> f32, double-buffered LDS, async DMA staging
//   per wave: 64x64 tile = 4x4 accumulators, 32 WMMA / 32 ds_load per stage
// =======================================================================
__launch_bounds__(256)
__global__ void gemm_bf16_wmma(const __bf16* __restrict__ xb,
                               const __bf16* __restrict__ wb,
                               const float* __restrict__ bias,
                               float* __restrict__ out,
                               int M, int N, int K)
{
    __shared__ __bf16 As[2][BM * LDA];   // x tile  [128 rows][BK]
    __shared__ __bf16 Ws[2][BN * LDB];   // w tile  [256 cols][BK]

    const int tid  = threadIdx.x;
    const int lane = tid & 31;
    const int wave = tid >> 5;
    const int waveRow = wave >> 2;   // 0..1 -> 64 rows
    const int waveCol = wave & 3;    // 0..3 -> 64 cols

    const int m0 = blockIdx.y * BM;
    const int n0 = blockIdx.x * BN;

    v8f acc[4][4];
    const v8f zero = {};
#pragma unroll
    for (int tn = 0; tn < 4; ++tn)
#pragma unroll
        for (int tm = 0; tm < 4; ++tm)
            acc[tn][tm] = zero;

    const int rA = lane & 15;
    const int kA = (lane < 16) ? 0 : 8;    // 16-bit A-matrix lane K base
    const int kB = (lane < 16) ? 0 : 8;    // 16-bit B-matrix lane K base

    // ---- tile staging: 16B chunks; A: 1024 chunks (4/thr), B: 2048 (8/thr) ----
#if HAVE_ASYNC_LDS
    auto stage = [&](int p, int k0) {
#pragma unroll
        for (int i = 0; i < 4; ++i) {
            int f = tid + i * 256;          // A chunk id
            int r = f >> 3, c = f & 7;      // 8 chunks per 64-elem row
            async_cp_b128(xb + (size_t)(m0 + r) * K + k0 + c * 8,
                          &As[p][r * LDA + c * 8]);
        }
#pragma unroll
        for (int i = 0; i < 8; ++i) {
            int f = tid + i * 256;          // B chunk id
            int r = f >> 3, c = f & 7;
            async_cp_b128(wb + (size_t)(n0 + r) * K + k0 + c * 8,
                          &Ws[p][r * LDB + c * 8]);
        }
    };
#else
    auto stage = [&](int p, int k0) {
        v8bf tx[4], tw[8];
#pragma unroll
        for (int i = 0; i < 4; ++i) {
            int f = tid + i * 256; int r = f >> 3, c = f & 7;
            tx[i] = *(const v8bf*)(xb + (size_t)(m0 + r) * K + k0 + c * 8);
        }
#pragma unroll
        for (int i = 0; i < 8; ++i) {
            int f = tid + i * 256; int r = f >> 3, c = f & 7;
            tw[i] = *(const v8bf*)(wb + (size_t)(n0 + r) * K + k0 + c * 8);
        }
#pragma unroll
        for (int i = 0; i < 4; ++i) {
            int f = tid + i * 256; int r = f >> 3, c = f & 7;
            *(v8bf*)&As[p][r * LDA + c * 8] = tx[i];
        }
#pragma unroll
        for (int i = 0; i < 8; ++i) {
            int f = tid + i * 256; int r = f >> 3, c = f & 7;
            *(v8bf*)&Ws[p][r * LDB + c * 8] = tw[i];
        }
    };
#endif

    auto compute = [&](int p) {
#pragma unroll
        for (int kc = 0; kc < 2; ++kc) {          // two 16x16x32 K-chunks
            v16bf afrag[4];
#pragma unroll
            for (int tm = 0; tm < 4; ++tm) {
                int row = waveRow * 64 + tm * 16 + rA;
                v8bf lo = *(const v8bf*)&As[p][row * LDA + kc * 32 + kA];
                v8bf hi = *(const v8bf*)&As[p][row * LDA + kc * 32 + 16 + kA];
                afrag[tm] = __builtin_shufflevector(lo, hi,
                    0,1,2,3,4,5,6,7,8,9,10,11,12,13,14,15);
            }
#pragma unroll
            for (int tn = 0; tn < 4; ++tn) {
                int col = waveCol * 64 + tn * 16 + rA;
                v8bf lo = *(const v8bf*)&Ws[p][col * LDB + kc * 32 + kB];
                v8bf hi = *(const v8bf*)&Ws[p][col * LDB + kc * 32 + kB + 16];
                v16bf bfrag = __builtin_shufflevector(lo, hi,
                    0,1,2,3,4,5,6,7,8,9,10,11,12,13,14,15);
#pragma unroll
                for (int tm = 0; tm < 4; ++tm)
                    acc[tn][tm] = __builtin_amdgcn_wmma_f32_16x16x32_bf16(
                        false, afrag[tm], false, bfrag,
                        (short)0, acc[tn][tm], false, false);
            }
        }
    };

    // ---- double-buffered main loop, last iteration peeled ----
    int p = 0;
    stage(0, 0);
    for (int k0 = 0; k0 < K - BK; k0 += BK, p ^= 1) {
        stage(p ^ 1, k0 + BK);
#if HAVE_ASYNC_LDS
        __builtin_amdgcn_s_wait_asynccnt(12);   // next tile's 12 DMAs in flight
#endif
        __syncthreads();
        compute(p);
        __syncthreads();
    }
#if HAVE_ASYNC_LDS
    __builtin_amdgcn_s_wait_asynccnt(0);
#endif
    __syncthreads();
    compute(p);

    // ---- epilogue: bias + f32 store (C/D: VGPR i -> M=i / M=i+8) ----
#pragma unroll
    for (int tn = 0; tn < 4; ++tn) {
        int col = n0 + waveCol * 64 + tn * 16 + (lane & 15);
        float bv = bias[col];
#pragma unroll
        for (int tm = 0; tm < 4; ++tm) {
            int rowBase = m0 + waveRow * 64 + tm * 16 + ((lane < 16) ? 0 : 8);
#pragma unroll
            for (int i = 0; i < 8; ++i) {
                out[(size_t)(rowBase + i) * N + col] = acc[tn][tm][i] + bv;
            }
        }
    }
}

// =======================================================================
// Fallback: fully fused (used only when d_ws < 96 MB)
// =======================================================================
__launch_bounds__(256, 2)
__global__ void gemm_fused_wmma(const float* __restrict__ x,
                                const int*   __restrict__ wq,
                                const float* __restrict__ scale_p,
                                const float* __restrict__ bias,
                                float* __restrict__ out,
                                int M, int N, int K)
{
    __shared__ __bf16 As[128 * 40];
    __shared__ __bf16 Ws[128 * 40];
    const int tid  = threadIdx.x;
    const int lane = tid & 31;
    const int wave = tid >> 5;
    const int waveRow = wave >> 2, waveCol = wave & 3;
    const int m0 = blockIdx.y * 128, n0 = blockIdx.x * 128;
    const float scale = scale_p[0];
    const int K4 = K >> 2;

    v8f acc[2][4];
    const v8f zero = {};
#pragma unroll
    for (int tn = 0; tn < 2; ++tn)
#pragma unroll
        for (int tm = 0; tm < 4; ++tm) acc[tn][tm] = zero;

    const int rA = lane & 15;
    const int kA = (lane < 16) ? 0 : 8;
    const int kB = (lane < 16) ? 0 : 8;

    for (int k0 = 0; k0 < K; k0 += 32) {
        float4 xv[4]; int4 wv[4];
        const float4* xs = (const float4*)(x + (size_t)m0 * K + k0);
        const int4*   ws = (const int4*)(wq + (size_t)n0 * K + k0);
#pragma unroll
        for (int i = 0; i < 4; ++i) {
            int f = tid + i * 256; int r = f >> 3, c = f & 7;
            xv[i] = xs[(size_t)r * K4 + c];
        }
#pragma unroll
        for (int i = 0; i < 4; ++i) {
            int f = tid + i * 256; int r = f >> 3, c = f & 7;
            wv[i] = ws[(size_t)r * K4 + c];
        }
#pragma unroll
        for (int i = 0; i < 4; ++i) {
            int f = tid + i * 256; int r = f >> 3, c = f & 7;
            __bf16* d = &As[r * 40 + c * 4];
            d[0] = (__bf16)xv[i].x; d[1] = (__bf16)xv[i].y;
            d[2] = (__bf16)xv[i].z; d[3] = (__bf16)xv[i].w;
        }
#pragma unroll
        for (int i = 0; i < 4; ++i) {
            int f = tid + i * 256; int r = f >> 3, c = f & 7;
            __bf16* d = &Ws[r * 40 + c * 4];
            d[0] = (__bf16)((float)wv[i].x * scale);
            d[1] = (__bf16)((float)wv[i].y * scale);
            d[2] = (__bf16)((float)wv[i].z * scale);
            d[3] = (__bf16)((float)wv[i].w * scale);
        }
        __syncthreads();
        v16bf afrag[4];
#pragma unroll
        for (int tm = 0; tm < 4; ++tm) {
            int row = waveRow * 64 + tm * 16 + rA;
            v8bf lo = *(const v8bf*)&As[row * 40 + kA];
            v8bf hi = *(const v8bf*)&As[row * 40 + 16 + kA];
            afrag[tm] = __builtin_shufflevector(lo, hi,
                0,1,2,3,4,5,6,7,8,9,10,11,12,13,14,15);
        }
        v16bf bfrag[2];
#pragma unroll
        for (int tn = 0; tn < 2; ++tn) {
            int col = waveCol * 32 + tn * 16 + rA;
            v8bf lo = *(const v8bf*)&Ws[col * 40 + kB];
            v8bf hi = *(const v8bf*)&Ws[col * 40 + kB + 16];
            bfrag[tn] = __builtin_shufflevector(lo, hi,
                0,1,2,3,4,5,6,7,8,9,10,11,12,13,14,15);
        }
#pragma unroll
        for (int tn = 0; tn < 2; ++tn)
#pragma unroll
            for (int tm = 0; tm < 4; ++tm)
                acc[tn][tm] = __builtin_amdgcn_wmma_f32_16x16x32_bf16(
                    false, afrag[tm], false, bfrag[tn],
                    (short)0, acc[tn][tm], false, false);
        __syncthreads();
    }
#pragma unroll
    for (int tn = 0; tn < 2; ++tn) {
        int col = n0 + waveCol * 32 + tn * 16 + (lane & 15);
        float bv = bias[col];
#pragma unroll
        for (int tm = 0; tm < 4; ++tm) {
            int rowBase = m0 + waveRow * 64 + tm * 16 + ((lane < 16) ? 0 : 8);
#pragma unroll
            for (int i = 0; i < 8; ++i)
                out[(size_t)(rowBase + i) * N + col] = acc[tn][tm][i] + bv;
        }
    }
}

extern "C" void kernel_launch(void* const* d_in, const int* in_sizes, int n_in,
                              void* d_out, int out_size, void* d_ws, size_t ws_size,
                              hipStream_t stream) {
    const float* x     = (const float*)d_in[0];   // (4,2048,4096) f32
    const int*   wq    = (const int*)  d_in[1];   // (4096,4096) int codes
    const float* scale = (const float*)d_in[2];   // (1,) f32
    const float* bias  = (const float*)d_in[3];   // (4096,) f32
    float*       out   = (float*)d_out;

    const int M = 4 * 2048, N = 4096, K = 4096;
    const size_t wbytes = (size_t)N * K * sizeof(__bf16);   // 32 MB
    const size_t xbytes = (size_t)M * K * sizeof(__bf16);   // 64 MB

    if (ws_size >= wbytes + xbytes) {
        __bf16* wb = (__bf16*)d_ws;
        __bf16* xb = (__bf16*)((char*)d_ws + wbytes);
        cvt_w_bf16<<<(N * (size_t)K) / (256 * 8), 256, 0, stream>>>(wq, scale, wb);
        cvt_x_bf16<<<((size_t)M * K) / (256 * 8), 256, 0, stream>>>(x, xb);
        dim3 grid(N / BN, M / BM);   // 16 x 64
        gemm_bf16_wmma<<<grid, 256, 0, stream>>>(xb, wb, bias, out, M, N, K);
    } else {
        dim3 grid(N / 128, M / 128); // 32 x 64
        gemm_fused_wmma<<<grid, 256, 0, stream>>>(x, wq, scale, bias, out, M, N, K);
    }
}
